// ForegroundAwarePatchFiltering_85736137163245
// MI455X (gfx1250) — matrix operand order repaired and checked by reading, not compiled, not run on hardware
//
#include <hip/hip_runtime.h>
#include <hip/hip_bf16.h>

// Problem sizes (fixed by reference)
#define B_  32
#define N_  4096
#define D_  768
#define H_  384
#define M_  (B_ * N_)          // 131072 rows of the big GEMM
#define KT  (D_ / 32)          // 24 K-steps of 32
#define HT  (H_ / 16)          // 24 output column tiles of 16
#define WAVES 2                // waves per workgroup in the score kernel
#define TPW   2                // 16-row M tiles per wave (B-fragment reuse factor)
#define LDS_STRIDE (D_ + 8)    // pad: 776 bf16 -> 1552B -> 388 dw -> bank-conflict-free

typedef __bf16 v16bf __attribute__((ext_vector_type(16)));
typedef float  v8f   __attribute__((ext_vector_type(8)));

typedef union { v16bf v; uint4 u[2]; } Frag;

__device__ __forceinline__ unsigned int f2bf_rne(float f) {
    unsigned int u = __float_as_uint(f);
    u += 0x7FFFu + ((u >> 16) & 1u);       // round-to-nearest-even
    return (u >> 16) & 0xFFFFu;
}

// ---------------------------------------------------------------------------
// Kernel 0: transpose + convert w1 [D,H] fp32 -> w1t [H,D] bf16 (576 KB, L2)
// ---------------------------------------------------------------------------
__global__ __launch_bounds__(256)
void k_prep_w1t(const float* __restrict__ w1, unsigned short* __restrict__ w1t) {
    int idx = blockIdx.x * 256 + threadIdx.x;           // over H_*D_
    if (idx >= H_ * D_) return;
    int h = idx / D_;
    int d = idx - h * D_;
    w1t[idx] = (unsigned short)f2bf_rne(w1[d * H_ + h]);
}

// ---------------------------------------------------------------------------
// Kernel 1: fused score head.  One wave = two 16-row M tiles (32 tokens):
// every B fragment feeds two WMMAs (halved WGP$/L2 traffic), and B is
// software-pipelined (double buffer, prefetch wraps across the t boundary)
// so a load pair is always in flight behind the matrix pipe.
//   h = GELU(x @ W1 + b1);  logit = h @ w2 + b2;  score = sigmoid(logit)
// ---------------------------------------------------------------------------
__global__ __launch_bounds__(32 * WAVES)
void k_scores(const float* __restrict__ x,
              const unsigned short* __restrict__ w1t,
              const float* __restrict__ b1,
              const float* __restrict__ w2,
              const float* __restrict__ b2,
              float* __restrict__ scores_out) {
    __shared__ unsigned short lds[WAVES * TPW * 16 * LDS_STRIDE];   // ~97 KB

    const int lane = threadIdx.x & 31;
    const int wv   = threadIdx.x >> 5;
    const int m0   = (blockIdx.x * WAVES + wv) * (16 * TPW);   // 32-row block
    unsigned short* ldsw = lds + wv * (TPW * 16 * LDS_STRIDE);

    // ---- Stage 1: load 32 x 768 fp32, convert to bf16 in LDS (wave-private)
    const float* xw = x + (size_t)m0 * D_;
    for (int r = 0; r < 16 * TPW; ++r) {
        const float* xr = xw + r * D_;
        unsigned short* lr = ldsw + r * LDS_STRIDE;
        for (int i = lane * 4; i < D_; i += 32 * 4) {
            float4 f = *(const float4*)(xr + i);
            unsigned int p0 = f2bf_rne(f.x) | (f2bf_rne(f.y) << 16);
            unsigned int p1 = f2bf_rne(f.z) | (f2bf_rne(f.w) << 16);
            *(uint2*)(lr + i) = make_uint2(p0, p1);
        }
    }
    asm volatile("s_wait_dscnt 0" ::: "memory");   // wave-private: no barrier needed

    // ---- Stage 2: WMMA over K; one shared B fragment drives both M tiles.
    const int  row  = lane & 15;     // A-fragment row this lane feeds; also C's N index
    const int  hsel = lane >> 4;     // 0: K 0..7,16..23 / M 0..7 ; 1: K 8..15,24..31 / M 8..15
    const char* ldsb0 = (const char*)ldsw;                               // tile 0
    const char* ldsb1 = (const char*)(ldsw + 16 * LDS_STRIDE);           // tile 1
    const int  kbsel = 8 * hsel;     // B K-offset for this half-wave

    float logit0[8], logit1[8];
#pragma unroll
    for (int j = 0; j < 8; ++j) { logit0[j] = 0.0f; logit1[j] = 0.0f; }

    // B double buffer; preload (t=0, s=0)
    Frag bbuf[2];
    {
        const char* wb0 = (const char*)w1t + (size_t)2 * row * D_;
        bbuf[0].u[0] = *(const uint4*)(wb0 + 2 * kbsel);
        bbuf[0].u[1] = *(const uint4*)(wb0 + 2 * kbsel + 32);
    }

    for (int t = 0; t < HT; ++t) {
        const int col = t * 16 + row;
        const char* wb = (const char*)w1t + (size_t)2 * col * D_;
        v8f c0 = {};
        v8f c1 = {};
#pragma unroll
        for (int s = 0; s < KT; ++s) {
            // Prefetch the next B fragment (wrapping to (t+1, s=0) at the end
            // of the K loop, so the epilogue hides the t-boundary latency).
            if (s + 1 < KT) {
                const int kbn = (s + 1) * 32 + kbsel;
                bbuf[(s + 1) & 1].u[0] = *(const uint4*)(wb + 2 * kbn);
                bbuf[(s + 1) & 1].u[1] = *(const uint4*)(wb + 2 * kbn + 32);
            } else if (t + 1 < HT) {
                const char* wbn = wb + (size_t)2 * 16 * D_;   // next column tile
                bbuf[0].u[0] = *(const uint4*)(wbn + 2 * kbsel);
                bbuf[0].u[1] = *(const uint4*)(wbn + 2 * kbsel + 32);
            }
            // A (16x32 bf16): lane<16 -> K k0+0..7 & k0+16..23 ; lane>=16 -> +8
            const int abase = row * LDS_STRIDE + s * 32 + 8 * hsel;
            Frag a0, a1;
            a0.u[0] = *(const uint4*)(ldsb0 + 2 * abase);
            a0.u[1] = *(const uint4*)(ldsb0 + 2 * abase + 32);
            a1.u[0] = *(const uint4*)(ldsb1 + 2 * abase);
            a1.u[1] = *(const uint4*)(ldsb1 + 2 * abase + 32);
            c0 = __builtin_amdgcn_wmma_f32_16x16x32_bf16(false, a0.v, false, bbuf[s & 1].v,
                                                         (short)0, c0, false, false);
            c1 = __builtin_amdgcn_wmma_f32_16x16x32_bf16(false, a1.v, false, bbuf[s & 1].v,
                                                         (short)0, c1, false, false);
        }
        // Epilogue: C layout -> VGPR j, this lane: M = j + 8*hsel, N = row (= col)
        const float bias = b1[col];
        const float wv2  = w2[col];
#pragma unroll
        for (int j = 0; j < 8; ++j) {
            float h0 = c0[j] + bias;
            float h1 = c1[j] + bias;
            float g0 = 0.5f * h0 * (1.0f + erff(h0 * 0.70710678118654752f));
            float g1 = 0.5f * h1 * (1.0f + erff(h1 * 0.70710678118654752f));
            logit0[j] += g0 * wv2;
            logit1[j] += g1 * wv2;
        }
    }

    // Reduce logit partials across the 16 lanes of each half (N dimension)
#pragma unroll
    for (int j = 0; j < 8; ++j) {
        float v0 = logit0[j], v1 = logit1[j];
        v0 += __shfl_xor(v0, 1, 32);  v1 += __shfl_xor(v1, 1, 32);
        v0 += __shfl_xor(v0, 2, 32);  v1 += __shfl_xor(v1, 2, 32);
        v0 += __shfl_xor(v0, 4, 32);  v1 += __shfl_xor(v1, 4, 32);
        v0 += __shfl_xor(v0, 8, 32);  v1 += __shfl_xor(v1, 8, 32);
        logit0[j] = v0;               logit1[j] = v1;
    }
    if (row == 0) {                         // lanes 0 and 16 write 8 rows per tile
        const float bb = b2[0];
#pragma unroll
        for (int j = 0; j < 8; ++j) {
            int r0 = m0 + j + 8 * hsel;
            int r1 = m0 + 16 + j + 8 * hsel;
            scores_out[r0] = 1.0f / (1.0f + expf(-(logit0[j] + bb)));
            scores_out[r1] = 1.0f / (1.0f + expf(-(logit1[j] + bb)));
        }
    }
}

// ---------------------------------------------------------------------------
// Kernel 2: per-batch exact k-th largest (bit-pattern bisection on the
// monotone count function), then mask + deterministic normalized weights.
// ---------------------------------------------------------------------------
__global__ __launch_bounds__(256)
void k_select(const float* __restrict__ scores, const int* __restrict__ pct_p,
              float* __restrict__ weights, float* __restrict__ mask_out) {
    __shared__ float s_lds[N_];
    __shared__ float red[256];
    __shared__ int   cnt_sh;

    const int b = blockIdx.x;
    const int tid = threadIdx.x;
    const float* sc = scores + (size_t)b * N_;
    for (int i = tid; i < N_; i += 256) s_lds[i] = sc[i];
    __syncthreads();

    int k = N_ * pct_p[0] / 100;
    if (k < 1) k = 1;
    if (k > N_) k = N_;

    float tau = -1.0f;                 // keep-all default (scores > 0)
    if (k < N_) {
        unsigned lo = 0u, hi = 0x7F800000u;      // positive-float bit ordering
        while (hi - lo > 1u) {
            unsigned mid = (lo + hi) >> 1;
            float mf = __uint_as_float(mid);
            if (tid == 0) cnt_sh = 0;
            __syncthreads();
            int c = 0;
            for (int i = tid; i < N_; i += 256) c += (s_lds[i] >= mf) ? 1 : 0;
            atomicAdd(&cnt_sh, c);               // exact int sum: deterministic
            __syncthreads();
            int total = cnt_sh;
            __syncthreads();
            if (total >= k) lo = mid; else hi = mid;   // uniform across block
        }
        tau = __uint_as_float(lo);               // = k-th largest score exactly
    }

    // mask, filtered scores, deterministic tree sum
    float psum = 0.0f;
    for (int i = tid; i < N_; i += 256) {
        float s = s_lds[i];
        float m = (k == N_) ? 1.0f : ((s >= tau) ? 1.0f : 0.0f);
        float f = s * m;
        mask_out[(size_t)b * N_ + i] = m;
        s_lds[i] = f;
        psum += f;
    }
    red[tid] = psum;
    __syncthreads();
    for (int st = 128; st > 0; st >>= 1) {
        if (tid < st) red[tid] += red[tid + st];
        __syncthreads();
    }
    const float inv = 1.0f / (red[0] + 1e-12f);
    for (int i = tid; i < N_; i += 256)
        weights[(size_t)b * N_ + i] = s_lds[i] * inv;
}

// ---------------------------------------------------------------------------
// Kernel 3: partial weighted feature sums over 256-token chunks (no atomics).
// Skips the ~50% of rows with zero weight (branch is wave-uniform).
// ---------------------------------------------------------------------------
__global__ __launch_bounds__(256)
void k_wsum_partial(const float* __restrict__ x, const float* __restrict__ weights,
                    float* __restrict__ partial) {
    __shared__ float wbuf[256];
    const int b  = blockIdx.x;
    const int ch = blockIdx.y;            // 16 chunks of 256 tokens
    const int tid = threadIdx.x;
    const int n0 = ch * 256;

    wbuf[tid] = weights[(size_t)b * N_ + n0 + tid];
    __syncthreads();

    float a0 = 0.f, a1 = 0.f, a2 = 0.f;
    const float* xb = x + ((size_t)b * N_ + n0) * D_;
    for (int nn = 0; nn < 256; ++nn) {
        float w = wbuf[nn];
        if (w != 0.0f) {                  // uniform: mask removed half the tokens
            const float* xr = xb + (size_t)nn * D_;
            a0 += w * xr[tid];
            a1 += w * xr[tid + 256];
            a2 += w * xr[tid + 512];
        }
    }
    float* pr = partial + (size_t)(b * 16 + ch) * D_;
    pr[tid] = a0; pr[tid + 256] = a1; pr[tid + 512] = a2;
}

// Kernel 4: deterministic reduction of the 16 chunk partials.
__global__ __launch_bounds__(256)
void k_feat_reduce(const float* __restrict__ partial, float* __restrict__ feat) {
    int idx = blockIdx.x * 256 + threadIdx.x;     // over B_*D_
    if (idx >= B_ * D_) return;
    int b = idx / D_;
    int d = idx - b * D_;
    float s = 0.f;
    for (int ch = 0; ch < 16; ++ch)
        s += partial[(size_t)(b * 16 + ch) * D_ + d];
    feat[idx] = s;
}

// ---------------------------------------------------------------------------
extern "C" void kernel_launch(void* const* d_in, const int* in_sizes, int n_in,
                              void* d_out, int out_size, void* d_ws, size_t ws_size,
                              hipStream_t stream) {
    const float* x   = (const float*)d_in[0];   // [B,N,D]
    const float* w1  = (const float*)d_in[1];   // [D,H]
    const float* b1  = (const float*)d_in[2];   // [H]
    const float* w2  = (const float*)d_in[3];   // [H]
    const float* b2  = (const float*)d_in[4];   // []
    const int*   pct = (const int*)d_in[5];     // keep_ratio_pct

    float* out     = (float*)d_out;             // outputs concatenated flat
    float* feat    = out;                                   // [B,D]
    float* weights = out + B_ * D_;                         // [B,N]
    float* scores  = out + B_ * D_ + (size_t)B_ * N_;       // [B,N]
    float* masko   = out + B_ * D_ + (size_t)2 * B_ * N_;   // [B,N]

    // Workspace: [w1t bf16 : 589824 B][partials fp32 : 1572864 B] (~2.2 MB)
    unsigned short* w1t = (unsigned short*)d_ws;
    float* partial = (float*)((char*)d_ws + (size_t)H_ * D_ * 2);

    k_prep_w1t<<<(H_ * D_ + 255) / 256, 256, 0, stream>>>(w1, w1t);

    k_scores<<<M_ / (16 * TPW * WAVES), 32 * WAVES, 0, stream>>>(x, w1t, b1, w2, b2, scores);

    k_select<<<B_, 256, 0, stream>>>(scores, pct, weights, masko);

    k_wsum_partial<<<dim3(B_, 16), 256, 0, stream>>>(x, weights, partial);

    k_feat_reduce<<<(B_ * D_ + 255) / 256, 256, 0, stream>>>(partial, feat);
}
